// PeepholeLSTM_87789131530607
// MI455X (gfx1250) — compile-verified, matrix operand order checked by reading, not run on hardware
//
#include <hip/hip_runtime.h>
#include <hip/hip_bf16.h>

// ---------------------------------------------------------------------------
// Peephole-free LSTM, B=128 T=512 I=256 H=512, f16 WMMA + f32 accumulate.
// Persistent cooperative kernel: 64 blocks x 256 threads (512 wave32).
// ---------------------------------------------------------------------------

typedef _Float16 h8   __attribute__((ext_vector_type(8)));
typedef _Float16 v16h __attribute__((ext_vector_type(16)));
typedef float    v8f  __attribute__((ext_vector_type(8)));

#define B_   128
#define T_   512
#define I_   256
#define H_   512
#define KTOT 768            // I_ + H_  (concatenated [x_t | h] K-dim)
#define NBLK 64

// workspace layout (bytes), all regions 256B aligned
#define OFF_CNT 0
#define OFF_H16 256
#define OFF_C   (OFF_H16 + B_*H_*2)          // 131328
#define OFF_PRE (OFF_C   + B_*H_*4)          // 393472
#define OFF_W   (OFF_PRE + 4*B_*H_*4)        // 1442048
#define OFF_X   (OFF_W   + 4*H_*KTOT*2)      // 4587776
// total ~38.1 MB

// ---------------------------------------------------------------------------
// f32 -> f16 conversion of x, re-laid out time-major: x16[t][b][i]
// ---------------------------------------------------------------------------
__global__ void cvt_x_kernel(const float* __restrict__ x, _Float16* __restrict__ x16) {
    const int n = B_ * T_ * I_;
    for (int idx = blockIdx.x * blockDim.x + threadIdx.x; idx < n;
         idx += gridDim.x * blockDim.x) {
        int i = idx & (I_ - 1);
        int t = (idx >> 8) & (T_ - 1);   // I_ = 256
        int b = idx >> 17;               // T_*I_ = 2^17
        x16[((size_t)t * B_ + b) * I_ + i] = (_Float16)x[idx];
    }
}

// ---------------------------------------------------------------------------
// Build concatenated f16 weights: wcat[g][ho][k], k<256 -> U_g, else W_g
// gate order g: 0=i, 1=f, 2=o, 3=c
// ---------------------------------------------------------------------------
__global__ void cvt_w_kernel(const float* U0, const float* W0,
                             const float* U1, const float* W1,
                             const float* U2, const float* W2,
                             const float* U3, const float* W3,
                             _Float16* __restrict__ wcat) {
    const float* Us[4] = {U0, U1, U2, U3};
    const float* Ws[4] = {W0, W1, W2, W3};
    const int n = 4 * H_ * KTOT;
    for (int idx = blockIdx.x * blockDim.x + threadIdx.x; idx < n;
         idx += gridDim.x * blockDim.x) {
        int k  = idx % KTOT;
        int ho = (idx / KTOT) % H_;
        int g  = idx / (KTOT * H_);
        float v = (k < I_) ? Us[g][ho * I_ + k] : Ws[g][ho * H_ + (k - I_)];
        wcat[idx] = (_Float16)v;
    }
}

// ---------------------------------------------------------------------------
// Zero h0/c0 state and the grid-barrier counter (runs every launch/replay)
// ---------------------------------------------------------------------------
__global__ void init_kernel(_Float16* __restrict__ h16, float* __restrict__ cst,
                            unsigned* __restrict__ cnt) {
    int idx = blockIdx.x * blockDim.x + threadIdx.x;
    if (idx < B_ * H_) {
        h16[idx] = (_Float16)0.f;
        cst[idx] = 0.f;
    }
    if (idx == 0) *cnt = 0u;
}

// ---------------------------------------------------------------------------
// device-wide barrier (monotonic counter, no reset needed: 64*1024 < 2^32)
// ---------------------------------------------------------------------------
__device__ __forceinline__ void grid_sync(unsigned* cnt, unsigned* phase) {
    __syncthreads();
    ++(*phase);
    if (threadIdx.x == 0) {
        __threadfence();
        atomicAdd(cnt, 1u);
        const unsigned target = (unsigned)NBLK * (*phase);
        while (__hip_atomic_load(cnt, __ATOMIC_RELAXED, __HIP_MEMORY_SCOPE_AGENT) < target)
            __builtin_amdgcn_s_sleep(2);
        __threadfence();
    }
    __syncthreads();
}

// load a v16h fragment: two contiguous 16B chunks 16 halfs apart
// (matches 16-bit A/B matrix lane layout, ISA 7.12.2)
__device__ __forceinline__ v16h frag16(const _Float16* p) {
    h8 lo = *(const h8*)p;
    h8 hi = *(const h8*)(p + 16);
    return __builtin_shufflevector(lo, hi, 0, 1, 2, 3, 4, 5, 6, 7,
                                           8, 9, 10, 11, 12, 13, 14, 15);
}

// ---------------------------------------------------------------------------
// Persistent LSTM kernel.
//  - block owns 2 column-groups cg = blockIdx*2 + {0,1}; cg -> (g = cg>>5,
//    h-tile = cg&31).  Their 2x16x768 f16 weight slab lives in LDS (48KB).
//  - wave w handles tiles gt = (blockIdx*8+w)*2 + {0,1}  ->  one cg, two
//    batch-tiles.  48 v_wmma_f32_16x16x32_f16 per wave per timestep.
// ---------------------------------------------------------------------------
__global__ void __launch_bounds__(256, 1)
lstm_kernel(const _Float16* __restrict__ x16, const _Float16* __restrict__ wcat,
            _Float16* __restrict__ h16, float* __restrict__ cst,
            float* __restrict__ pre, unsigned* __restrict__ cnt,
            const float* __restrict__ bias_i, const float* __restrict__ bias_f,
            const float* __restrict__ bias_o, const float* __restrict__ bias_c,
            float* __restrict__ out) {
    __shared__ __align__(16) _Float16 wlds[2 * 16 * KTOT];   // 48 KB

    // stage this block's weight slab into LDS once (reused for all 512 steps)
    {
        const h8* src = (const h8*)(wcat + (size_t)blockIdx.x * 2 * 16 * KTOT);
        h8* dst = (h8*)wlds;
        for (int i = threadIdx.x; i < 2 * 16 * KTOT / 8; i += 256) dst[i] = src[i];
    }
    __syncthreads();

    const int lane = threadIdx.x & 31;
    const int wave = threadIdx.x >> 5;
    const int gt0  = (blockIdx.x * 8 + wave) * 2;
    const int cg   = gt0 >> 3;          // global column-group
    const int cgl  = cg & 1;            // local (0/1) within block
    const int g    = cg >> 5;           // gate
    const int ht   = cg & 31;           // h-tile
    const int bt0  = gt0 & 7;           // first batch tile (even)
    const int r    = lane & 15;         // row (A) / col (B) / N (C)
    const int s    = (lane >> 4) << 3;  // 0 or 8: K-chunk select, also M offset

    const _Float16* wrow = &wlds[(size_t)(cgl * 16 + r) * KTOT];
    unsigned phase = 0;
    const int tid = blockIdx.x * 256 + threadIdx.x;

    for (int t = 0; t < T_; ++t) {
        v8f c0 = {}; v8f c1 = {};
        const _Float16* xrow0 = x16 + ((size_t)t * B_ + bt0 * 16 + r) * I_;
        const _Float16* xrow1 = xrow0 + 16 * I_;
        const _Float16* hrow0 = h16 + (size_t)(bt0 * 16 + r) * H_;
        const _Float16* hrow1 = hrow0 + 16 * H_;
        if (t + 1 < T_) __builtin_prefetch(xrow0 + (size_t)B_ * I_, 0, 0);

        // K = 0..255 : x_t @ U_g^T
#pragma unroll
        for (int k0 = 0; k0 < I_; k0 += 32) {
            v16h bfrag = frag16(wrow + k0 + s);
            v16h a0 = frag16(xrow0 + k0 + s);
            v16h a1 = frag16(xrow1 + k0 + s);
            c0 = __builtin_amdgcn_wmma_f32_16x16x32_f16(false, a0, false, bfrag,
                                                        (short)0, c0, false, false);
            c1 = __builtin_amdgcn_wmma_f32_16x16x32_f16(false, a1, false, bfrag,
                                                        (short)0, c1, false, false);
        }
        // K = 256..767 : h @ W_g^T
#pragma unroll
        for (int k0 = 0; k0 < H_; k0 += 32) {
            v16h bfrag = frag16(wrow + I_ + k0 + s);
            v16h a0 = frag16(hrow0 + k0 + s);
            v16h a1 = frag16(hrow1 + k0 + s);
            c0 = __builtin_amdgcn_wmma_f32_16x16x32_f16(false, a0, false, bfrag,
                                                        (short)0, c0, false, false);
            c1 = __builtin_amdgcn_wmma_f32_16x16x32_f16(false, a1, false, bfrag,
                                                        (short)0, c1, false, false);
        }

        // store preactivations: pre[g][b][ho]; C/D layout M=v+s, N=r
        float* p0 = pre + ((size_t)g * B_ + bt0 * 16 + s) * H_ + ht * 16 + r;
        float* p1 = p0 + 16 * H_;
#pragma unroll
        for (int v = 0; v < 8; ++v) {
            p0[(size_t)v * H_] = c0[v];
            p1[(size_t)v * H_] = c1[v];
        }

        grid_sync(cnt, &phase);   // all 4 gate preacts visible

        // elementwise LSTM cell update (65536 elems over 16384 threads)
        for (int idx = tid; idx < B_ * H_; idx += NBLK * 256) {
            int b  = idx >> 9;
            int ho = idx & (H_ - 1);
            float pi = pre[(size_t)(0 * B_ + b) * H_ + ho] + bias_i[ho];
            float pf = pre[(size_t)(1 * B_ + b) * H_ + ho] + bias_f[ho];
            float po = pre[(size_t)(2 * B_ + b) * H_ + ho] + bias_o[ho];
            float pc = pre[(size_t)(3 * B_ + b) * H_ + ho] + bias_c[ho];
            float ig = 1.f / (1.f + __expf(-pi));
            float fg = 1.f / (1.f + __expf(-pf));
            float og = 1.f / (1.f + __expf(-po));
            float cc = 2.f / (1.f + __expf(-2.f * pc)) - 1.f;      // tanh
            float cn = fg * cst[idx] + ig * cc;
            float th = 2.f / (1.f + __expf(-2.f * cn)) - 1.f;      // tanh
            float hn = og * th;
            cst[idx] = cn;
            h16[idx] = (_Float16)hn;
            out[((size_t)b * T_ + t) * H_ + ho] = hn;              // hidden_seq [B,T,H]
            if (t == T_ - 1) {
                out[(size_t)B_ * T_ * H_ + idx]            = hn;   // h_T
                out[(size_t)B_ * T_ * H_ + B_ * H_ + idx]  = cn;   // c_T
            }
        }

        grid_sync(cnt, &phase);   // h16 ready / preact free for next step
    }
}

// ---------------------------------------------------------------------------
extern "C" void kernel_launch(void* const* d_in, const int* in_sizes, int n_in,
                              void* d_out, int out_size, void* d_ws, size_t ws_size,
                              hipStream_t stream) {
    const float* x   = (const float*)d_in[0];
    const float* W_i = (const float*)d_in[1];
    const float* U_i = (const float*)d_in[2];
    const float* b_i = (const float*)d_in[3];
    const float* W_f = (const float*)d_in[4];
    const float* U_f = (const float*)d_in[5];
    const float* b_f = (const float*)d_in[6];
    const float* W_o = (const float*)d_in[7];
    const float* U_o = (const float*)d_in[8];
    const float* b_o = (const float*)d_in[9];
    const float* W_c = (const float*)d_in[10];
    const float* U_c = (const float*)d_in[11];
    const float* b_c = (const float*)d_in[12];

    char* ws = (char*)d_ws;
    unsigned*  cnt  = (unsigned*)(ws + OFF_CNT);
    _Float16*  h16  = (_Float16*)(ws + OFF_H16);
    float*     cst  = (float*)(ws + OFF_C);
    float*     pre  = (float*)(ws + OFF_PRE);
    _Float16*  wcat = (_Float16*)(ws + OFF_W);
    _Float16*  x16  = (_Float16*)(ws + OFF_X);

    cvt_x_kernel<<<4096, 256, 0, stream>>>(x, x16);
    cvt_w_kernel<<<1536, 256, 0, stream>>>(U_i, W_i, U_f, W_f, U_o, W_o, U_c, W_c, wcat);
    init_kernel<<<256, 256, 0, stream>>>(h16, cst, cnt);
    lstm_kernel<<<NBLK, 256, 0, stream>>>(x16, wcat, h16, cst, pre, cnt,
                                          b_i, b_f, b_o, b_c, (float*)d_out);
}
